// GTN_23794118820143
// MI455X (gfx1250) — compile-verified, matrix operand order checked by reading, not run on hardware
//
#include <hip/hip_runtime.h>
#include <hip/hip_bf16.h>
#include <math.h>
#include <stdint.h>

// Problem constants (fixed by the reference).
#define NN 2048
#define CC 4
#define EE 4
#define KIN 256
#define FF 128

#define KSTEP 32
#define LDL 40   // padded LDS row stride (elements) for bank-conflict-free frags

typedef __attribute__((ext_vector_type(16))) __bf16 v16bf;
typedef __attribute__((ext_vector_type(8)))  float  v8f;
typedef __attribute__((ext_vector_type(4)))  unsigned int v4u;
typedef int gtn_i32x4 __attribute__((vector_size(16)));

#if __has_builtin(__builtin_amdgcn_global_load_async_to_lds_b128)
#define GTN_HAS_ASYNC 1
#else
#define GTN_HAS_ASYNC 0
#endif

#if GTN_HAS_ASYNC
typedef __attribute__((address_space(1))) gtn_i32x4 gtn_g_i32x4;  // global
typedef __attribute__((address_space(3))) gtn_i32x4 gtn_l_i32x4;  // LDS
#endif

// ---------------------------------------------------------------------------
// WMMA bf16 fragment load from an LDS tile (row-major, row stride LDL elems).
// Per ISA 16-bit operand layout: lane L (L&15 = row), half h = L>>4:
//   elements 0..7  at K = 8h .. 8h+7      (16 contiguous bytes)
//   elements 8..15 at K = 16+8h .. 16+8h+7
// ---------------------------------------------------------------------------
__device__ __forceinline__ v16bf gtn_lds_frag(const __bf16* tile, int row,
                                              int lane) {
  const int r  = row + (lane & 15);
  const int kh = (lane >> 4) & 1;
  const __bf16* p = tile + r * LDL + 8 * kh;
  union { v4u u[2]; v16bf b; } f;
  f.u[0] = *(const v4u*)(p);
  f.u[1] = *(const v4u*)(p + 16);
  return f.b;
}

// ---------------------------------------------------------------------------
// Stage a 128x32 bf16 tile (rows [row0,row0+128) x cols [k0,k0+32) of a
// row-major matrix with row stride `ld`) into LDS dst[128][LDL].
// 512 chunks of 16B; 256 threads x 2 chunks. Async (ASYNCcnt) when available.
// ---------------------------------------------------------------------------
__device__ __forceinline__ void gtn_stage_tile(const __bf16* gbase, int row0,
                                               int k0, int ld,
                                               __bf16* dst, int tid) {
#pragma unroll
  for (int j = 0; j < 2; ++j) {
    const int q   = tid + 256 * j;   // 0..511
    const int row = q >> 2;          // 0..127
    const int co  = (q & 3) * 8;     // element offset in row: 0,8,16,24
    const __bf16* src = gbase + (size_t)(row0 + row) * ld + (k0 + co);
    __bf16* d = dst + row * LDL + co;
#if GTN_HAS_ASYNC
    __builtin_amdgcn_global_load_async_to_lds_b128(
        (gtn_g_i32x4*)(uintptr_t)src,
        (gtn_l_i32x4*)(unsigned)(uintptr_t)d,
        0, 0);
#else
    *(v4u*)d = *(const v4u*)src;
#endif
  }
}

__device__ __forceinline__ void gtn_wait_async() {
#if GTN_HAS_ASYNC
#if __has_builtin(__builtin_amdgcn_s_wait_asynccnt)
  __builtin_amdgcn_s_wait_asynccnt(0);
#else
  asm volatile("s_wait_asynccnt 0" ::: "memory");
#endif
#endif
}

// ---------------------------------------------------------------------------
// K0: row-softmax of gt_w1/gt_w2/gt_w3 (each [4,4]) -> fw[48]
// ---------------------------------------------------------------------------
__global__ void gtn_softmax_kernel(const float* __restrict__ w1,
                                   const float* __restrict__ w2,
                                   const float* __restrict__ w3,
                                   float* __restrict__ fw) {
  int t = threadIdx.x;
  if (t < 3 * CC) {
    const float* src = (t < CC) ? (w1 + t * EE)
                     : (t < 2 * CC) ? (w2 + (t - CC) * EE)
                     : (w3 + (t - 2 * CC) * EE);
    float mx = src[0];
    for (int e = 1; e < EE; ++e) mx = fmaxf(mx, src[e]);
    float v[EE], s = 0.f;
    for (int e = 0; e < EE; ++e) { v[e] = __expf(src[e] - mx); s += v[e]; }
    float inv = 1.f / s;
    for (int e = 0; e < EE; ++e) fw[t * EE + e] = v[e] * inv;
  }
}

// ---------------------------------------------------------------------------
// K1: Ha = sum_e fw1[c,e]*A[e] (bf16 row-major)
//     HbT/HcT = transposed bf16 mixes (LDS tile transpose, coalesced stores)
// grid (NN/32, NN/32, CC), block (32, 8)
// ---------------------------------------------------------------------------
__global__ void gtn_mix_kernel(const float* __restrict__ A,
                               const float* __restrict__ fw,
                               __bf16* __restrict__ Ha,
                               __bf16* __restrict__ HbT,
                               __bf16* __restrict__ HcT) {
  __shared__ __bf16 tb[32][33];
  __shared__ __bf16 tc[32][33];
  const int c  = blockIdx.z;
  const int m0 = blockIdx.x * 32;   // column tile
  const int n0 = blockIdx.y * 32;   // row tile
  float w1[EE], w2[EE], w3[EE];
#pragma unroll
  for (int e = 0; e < EE; ++e) {
    w1[e] = fw[c * EE + e];
    w2[e] = fw[16 + c * EE + e];
    w3[e] = fw[32 + c * EE + e];
  }
  const size_t mat = (size_t)NN * NN;
#pragma unroll
  for (int i = 0; i < 4; ++i) {
    const int lr  = threadIdx.y + 8 * i;     // local row
    const int row = n0 + lr;
    const int col = m0 + threadIdx.x;
    const size_t idx = (size_t)row * NN + col;
    float ha = 0.f, hb = 0.f, hc = 0.f;
#pragma unroll
    for (int e = 0; e < EE; ++e) {
      float v = A[(size_t)e * mat + idx];
      ha += w1[e] * v; hb += w2[e] * v; hc += w3[e] * v;
    }
    Ha[(size_t)c * mat + idx] = (__bf16)ha;
    tb[lr][threadIdx.x] = (__bf16)hb;
    tc[lr][threadIdx.x] = (__bf16)hc;
  }
  __syncthreads();
#pragma unroll
  for (int i = 0; i < 4; ++i) {
    const int lr   = threadIdx.y + 8 * i;    // local transposed row (= col idx)
    const int mrow = m0 + lr;
    const int ncol = n0 + threadIdx.x;
    const size_t o = (size_t)c * mat + (size_t)mrow * NN + ncol;
    HbT[o] = tb[threadIdx.x][lr];
    HcT[o] = tc[threadIdx.x][lr];
  }
}

// ---------------------------------------------------------------------------
// K2: supportT[f][m] = (X @ gcn_w)[m][f] in bf16 (B-operand of final matmul)
// ---------------------------------------------------------------------------
__global__ void gtn_support_kernel(const float* __restrict__ X,
                                   const float* __restrict__ W,
                                   __bf16* __restrict__ ST) {
  const int idx = blockIdx.x * blockDim.x + threadIdx.x;
  if (idx >= NN * FF) return;
  const int m = idx / FF, f = idx % FF;
  float s = 0.f;
  for (int k = 0; k < KIN; ++k) s += X[m * KIN + k] * W[k * FF + f];
  ST[(size_t)f * NN + m] = (__bf16)s;
}

// ---------------------------------------------------------------------------
// K3/K4: batched NxN @ NxN via v_wmma_f32_16x16x32_bf16 with double-buffered
// async LDS staging. Block = 256 threads = 8 waves; 128x128 tile; each wave
// 64 rows x 32 cols (4x2 accumulators). B pre-transposed (BT row-major).
// grid (NN/128, NN/128, CC)
// ---------------------------------------------------------------------------
template <typename OutT>
__global__ void gtn_bmm_kernel(const __bf16* __restrict__ Am,
                               const __bf16* __restrict__ BTm,
                               OutT* __restrict__ Dm) {
  __shared__ __attribute__((aligned(16))) __bf16 As[2][128][LDL];
  __shared__ __attribute__((aligned(16))) __bf16 Bs[2][128][LDL];

  const int c = blockIdx.z;
  const size_t mat = (size_t)NN * NN;
  const __bf16* Ab = Am  + (size_t)c * mat;
  const __bf16* Bb = BTm + (size_t)c * mat;
  OutT*         Db = Dm  + (size_t)c * mat;
  const int tid  = threadIdx.x;
  const int lane = tid & 31;
  const int w    = tid >> 5;
  const int wm   = w & 1;        // 0..1  -> 64-row slab within the tile
  const int wn   = w >> 1;       // 0..3  -> 32-col slab within the tile
  const int rowA = blockIdx.y * 128;
  const int rowB = blockIdx.x * 128;

  v8f acc[4][2];
#pragma unroll
  for (int i = 0; i < 4; ++i)
#pragma unroll
    for (int j = 0; j < 2; ++j) acc[i][j] = (v8f)(0.f);

  // Prologue: stage slab 0.
  gtn_stage_tile(Ab, rowA, 0, NN, &As[0][0][0], tid);
  gtn_stage_tile(Bb, rowB, 0, NN, &Bs[0][0][0], tid);
  gtn_wait_async();
  __syncthreads();

  const int NK = NN / KSTEP;
  for (int kt = 0; kt < NK; ++kt) {
    const int cur = kt & 1;
    if (kt + 1 < NK) {  // overlap: stage slab kt+1 while computing slab kt
      gtn_stage_tile(Ab, rowA, (kt + 1) * KSTEP, NN, &As[cur ^ 1][0][0], tid);
      gtn_stage_tile(Bb, rowB, (kt + 1) * KSTEP, NN, &Bs[cur ^ 1][0][0], tid);
    }
    v16bf a[4], b[2];
#pragma unroll
    for (int i = 0; i < 4; ++i)
      a[i] = gtn_lds_frag(&As[cur][0][0], wm * 64 + 16 * i, lane);
#pragma unroll
    for (int j = 0; j < 2; ++j)
      b[j] = gtn_lds_frag(&Bs[cur][0][0], wn * 32 + 16 * j, lane);
#pragma unroll
    for (int i = 0; i < 4; ++i)
#pragma unroll
      for (int j = 0; j < 2; ++j)
        acc[i][j] = __builtin_amdgcn_wmma_f32_16x16x32_bf16(
            false, a[i], false, b[j], (short)0, acc[i][j], false, false);
    gtn_wait_async();
    __syncthreads();
  }

  const int lm = 8 * ((lane >> 4) & 1);
  const int ln = lane & 15;
#pragma unroll
  for (int i = 0; i < 4; ++i)
#pragma unroll
    for (int j = 0; j < 2; ++j)
#pragma unroll
      for (int r = 0; r < 8; ++r) {
        const int mm = rowA + wm * 64 + 16 * i + r + lm;
        const int nn = rowB + wn * 32 + 16 * j + ln;
        Db[(size_t)mm * NN + nn] = (OutT)acc[i][j][r];
      }
}

// ---------------------------------------------------------------------------
// K5: d[c][n] = rsqrt(1 + sum_m H2[c][n][m])   (row sum of H + I)
// grid (NN, CC), block 256
// ---------------------------------------------------------------------------
__global__ void gtn_degree_kernel(const float* __restrict__ H2,
                                  float* __restrict__ dv) {
  const int c = blockIdx.y, n = blockIdx.x;
  const float* row = H2 + ((size_t)c * NN + n) * NN;
  float s = 0.f;
  for (int m = threadIdx.x; m < NN; m += 256) s += row[m];
  __shared__ float red[256];
  red[threadIdx.x] = s;
  __syncthreads();
  for (int off = 128; off > 0; off >>= 1) {
    if (threadIdx.x < off) red[threadIdx.x] += red[threadIdx.x + off];
    __syncthreads();
  }
  if (threadIdx.x == 0) {
    float deg = red[0] + 1.0f;
    dv[c * NN + n] = (deg > 0.f) ? rsqrtf(deg) : 0.f;
  }
}

// ---------------------------------------------------------------------------
// K6: Hn[c][n][m] = bf16( d[n] * (H2 + I) * d[m] )
// ---------------------------------------------------------------------------
__global__ void gtn_normalize_kernel(const float* __restrict__ H2,
                                     const float* __restrict__ dv,
                                     __bf16* __restrict__ Hn) {
  const unsigned idx = blockIdx.x * 256u + threadIdx.x;  // < CC*NN*NN = 2^24
  const unsigned c = idx >> 22;                          // NN*NN = 2^22
  const unsigned rem = idx & ((1u << 22) - 1u);
  const unsigned n = rem >> 11;                          // NN = 2^11
  const unsigned m = rem & (NN - 1);
  float v = H2[idx] + ((n == m) ? 1.f : 0.f);
  Hn[idx] = (__bf16)(dv[c * NN + n] * v * dv[c * NN + m]);
}

// ---------------------------------------------------------------------------
// K7: out[node][c*FF + f] = relu( (Hn[c] @ support)[node][f] + bias[f] )
// grid (NN/128, 1, CC), block 256 (8 waves), tile 128(M) x 128(F),
// double-buffered async LDS staging as in the bmm kernels.
// ---------------------------------------------------------------------------
__global__ void gtn_gcn_mm_kernel(const __bf16* __restrict__ Hn,
                                  const __bf16* __restrict__ ST,
                                  const float* __restrict__ bias,
                                  float* __restrict__ out) {
  __shared__ __attribute__((aligned(16))) __bf16 As[2][128][LDL];
  __shared__ __attribute__((aligned(16))) __bf16 Bs[2][128][LDL];

  const int c = blockIdx.z;
  const __bf16* Ab = Hn + (size_t)c * NN * NN;
  const int tid  = threadIdx.x;
  const int lane = tid & 31;
  const int w    = tid >> 5;
  const int wm   = w & 1;
  const int wn   = w >> 1;
  const int rowA = blockIdx.x * 128;

  v8f acc[4][2];
#pragma unroll
  for (int i = 0; i < 4; ++i)
#pragma unroll
    for (int j = 0; j < 2; ++j) acc[i][j] = (v8f)(0.f);

  gtn_stage_tile(Ab, rowA, 0, NN, &As[0][0][0], tid);
  gtn_stage_tile(ST, 0,    0, NN, &Bs[0][0][0], tid);
  gtn_wait_async();
  __syncthreads();

  const int NK = NN / KSTEP;
  for (int kt = 0; kt < NK; ++kt) {
    const int cur = kt & 1;
    if (kt + 1 < NK) {
      gtn_stage_tile(Ab, rowA, (kt + 1) * KSTEP, NN, &As[cur ^ 1][0][0], tid);
      gtn_stage_tile(ST, 0,    (kt + 1) * KSTEP, NN, &Bs[cur ^ 1][0][0], tid);
    }
    v16bf a[4], b[2];
#pragma unroll
    for (int i = 0; i < 4; ++i)
      a[i] = gtn_lds_frag(&As[cur][0][0], wm * 64 + 16 * i, lane);
#pragma unroll
    for (int j = 0; j < 2; ++j)
      b[j] = gtn_lds_frag(&Bs[cur][0][0], wn * 32 + 16 * j, lane);
#pragma unroll
    for (int i = 0; i < 4; ++i)
#pragma unroll
      for (int j = 0; j < 2; ++j)
        acc[i][j] = __builtin_amdgcn_wmma_f32_16x16x32_bf16(
            false, a[i], false, b[j], (short)0, acc[i][j], false, false);
    gtn_wait_async();
    __syncthreads();
  }

  const int lm = 8 * ((lane >> 4) & 1);
  const int ln = lane & 15;
#pragma unroll
  for (int i = 0; i < 4; ++i)
#pragma unroll
    for (int j = 0; j < 2; ++j) {
      const int f = wn * 32 + 16 * j + ln;
      const float bf = bias[f];
#pragma unroll
      for (int r = 0; r < 8; ++r) {
        const int node = rowA + wm * 64 + 16 * i + r + lm;
        float v = acc[i][j][r] + bf;
        out[(size_t)node * (CC * FF) + c * FF + f] = (v > 0.f) ? v : 0.f;
      }
    }
}

// ---------------------------------------------------------------------------
// Host launcher. Workspace layout (bytes):
//   [0,   32MB) Ha (bf16)         -> later low half of H2 (f32)
//   [32,  64MB) HbT (bf16)        -> later high half of H2 (f32)
//   [64,  96MB) HcT (bf16)
//   [96, 128MB) H1 (bf16)         -> later Hn (bf16)
//   [128MB, +512KB) supportT (bf16)
//   [129MB, +192B)  fw
//   [129MB+4KB, +32KB) d vector
// ---------------------------------------------------------------------------
extern "C" void kernel_launch(void* const* d_in, const int* in_sizes, int n_in,
                              void* d_out, int out_size, void* d_ws, size_t ws_size,
                              hipStream_t stream) {
  const float* A     = (const float*)d_in[0];
  const float* X     = (const float*)d_in[1];
  const float* gt_w1 = (const float*)d_in[2];
  const float* gt_w2 = (const float*)d_in[3];
  const float* gt_w3 = (const float*)d_in[4];
  const float* gcn_w = (const float*)d_in[5];
  const float* gcn_b = (const float*)d_in[6];
  float* out = (float*)d_out;

  char* ws = (char*)d_ws;
  const size_t MB = 1024u * 1024u;
  __bf16* Ha  = (__bf16*)(ws + 0 * MB);
  __bf16* HbT = (__bf16*)(ws + 32 * MB);
  __bf16* HcT = (__bf16*)(ws + 64 * MB);
  __bf16* H1  = (__bf16*)(ws + 96 * MB);
  float*  H2  = (float*)(ws + 0 * MB);      // reuses Ha+HbT after bmm1
  __bf16* Hn  = (__bf16*)(ws + 96 * MB);    // reuses H1 after bmm2
  __bf16* ST  = (__bf16*)(ws + 128 * MB);
  float*  fw  = (float*)(ws + 129 * MB);
  float*  dv  = (float*)(ws + 129 * MB + 4096);

  gtn_softmax_kernel<<<1, 32, 0, stream>>>(gt_w1, gt_w2, gt_w3, fw);
  gtn_mix_kernel<<<dim3(NN / 32, NN / 32, CC), dim3(32, 8), 0, stream>>>(
      A, fw, Ha, HbT, HcT);
  gtn_support_kernel<<<(NN * FF) / 256, 256, 0, stream>>>(X, gcn_w, ST);
  gtn_bmm_kernel<__bf16><<<dim3(NN / 128, NN / 128, CC), 256, 0, stream>>>(
      Ha, HbT, H1);
  gtn_bmm_kernel<float><<<dim3(NN / 128, NN / 128, CC), 256, 0, stream>>>(
      H1, HcT, H2);
  gtn_degree_kernel<<<dim3(NN, CC), 256, 0, stream>>>(H2, dv);
  gtn_normalize_kernel<<<(CC * NN * NN) / 256, 256, 0, stream>>>(H2, dv, Hn);
  gtn_gcn_mm_kernel<<<dim3(NN / 128, 1, CC), 256, 0, stream>>>(Hn, ST, gcn_b, out);
}
